// GraphSAGE_44461501448672
// MI455X (gfx1250) — matrix operand order repaired and checked by reading, not compiled, not run on hardware
//
#include <hip/hip_runtime.h>
#include <hip/hip_bf16.h>

typedef __attribute__((ext_vector_type(16))) __bf16 v16bf;
typedef __attribute__((ext_vector_type(8)))  float  v8f;

#define IN_CH  100
#define HID_CH 128
#define OUT_CH 47

// ---------------------------------------------------------------- helpers

__device__ __forceinline__ v8f wmma_bf16(v16bf a, v16bf b, v8f c) {
  // 8 args: (neg_a, A, neg_b, B, c_mod, C, reuse_a, reuse_b)
  return __builtin_amdgcn_wmma_f32_16x16x32_bf16(false, a, false, b,
                                                 (short)0, c, false, false);
}

// split fp32 -> bf16 hi + bf16 lo residual (hi+lo ~= fp32)
__device__ __forceinline__ void split16(const float* f, v16bf& hi, v16bf& lo) {
#pragma unroll
  for (int j = 0; j < 16; ++j) {
    __bf16 h = (__bf16)f[j];
    hi[j] = h;
    lo[j] = (__bf16)(f[j] - (float)h);
  }
}

// guarded float4 load with zero padding past kvalid
__device__ __forceinline__ float4 ld4g(const float* __restrict__ row, int k, int kvalid) {
  float4 v = make_float4(0.f, 0.f, 0.f, 0.f);
  if (k + 4 <= kvalid) {
    v = *(const float4*)(row + k);
  } else {
    if (k     < kvalid) v.x = row[k];
    if (k + 1 < kvalid) v.y = row[k + 1];
    if (k + 2 < kvalid) v.z = row[k + 2];
    if (k + 3 < kvalid) v.w = row[k + 3];
  }
  return v;
}

// Build one per-lane A fragment (16 bf16 hi + 16 bf16 lo) for WMMA 16x16x32.
// A layout (16-bit A, wave32): lane half h holds K = kb..kb+7 and kb+16..kb+23
// where kb = kstep_base + h*8 (kb passed in already including h*8).
__device__ __forceinline__ void build_frag(const float* __restrict__ srow, int kb,
                                           int kvalid, float scl,
                                           v16bf& hi, v16bf& lo) {
  float4 f0 = ld4g(srow, kb,      kvalid);
  float4 f1 = ld4g(srow, kb + 4,  kvalid);
  float4 f2 = ld4g(srow, kb + 16, kvalid);
  float4 f3 = ld4g(srow, kb + 20, kvalid);
  float f[16] = { f0.x*scl, f0.y*scl, f0.z*scl, f0.w*scl,
                  f1.x*scl, f1.y*scl, f1.z*scl, f1.w*scl,
                  f2.x*scl, f2.y*scl, f2.z*scl, f2.w*scl,
                  f3.x*scl, f3.y*scl, f3.z*scl, f3.w*scl };
  split16(f, hi, lo);
}

// ------------------------------------------------- weight pre-swizzle (once)
// B-fragment layout per wave32 WMMA 16x16x32 bf16:
//   lanes 0-15 : N = lane,    K = 0..7, 16..23  (2 K per VGPR)
//   lanes 16-31: N = lane-16, K = 8..15, 24..31
// We store per (ntile, kstep, lane) a contiguous run of 16 bf16 values.
// Combined K axis: ksteps 0..3 = W_l (K 0..127), ksteps 4..7 = W_r.
__global__ void prep_weights(const float* __restrict__ w1l, const float* __restrict__ w1r,
                             const float* __restrict__ w2l, const float* __restrict__ w2r,
                             __bf16* __restrict__ bw1hi, __bf16* __restrict__ bw1lo,
                             __bf16* __restrict__ bw2hi, __bf16* __restrict__ bw2lo) {
  int t = blockIdx.x * blockDim.x + threadIdx.x;
  if (t < 8 * 8 * 32) {                       // layer 1: 8 ntiles x 8 ksteps x 32 lanes
    int lane = t & 31, ks = (t >> 5) & 7, nt = t >> 8;
    int n = nt * 16 + (lane & 15);
    int kofs = (lane >> 4) << 3;
    int base = t * 16;
#pragma unroll
    for (int j = 0; j < 16; ++j) {
      int k = (ks & 3) * 32 + kofs + (j < 8 ? j : j + 8);       // source-local K (padded 100->128)
      float v = 0.f;
      if (k < IN_CH) v = (ks < 4) ? w1l[k * HID_CH + n] : w1r[k * HID_CH + n];
      __bf16 h = (__bf16)v;
      bw1hi[base + j] = h;
      bw1lo[base + j] = (__bf16)(v - (float)h);
    }
  } else if (t < 8 * 8 * 32 + 3 * 8 * 32) {   // layer 2: 3 ntiles x 8 ksteps x 32 lanes
    int u = t - 8 * 8 * 32;
    int lane = u & 31, ks = (u >> 5) & 7, nt = u >> 8;
    int n = nt * 16 + (lane & 15);
    int kofs = (lane >> 4) << 3;
    int base = u * 16;
#pragma unroll
    for (int j = 0; j < 16; ++j) {
      int k = (ks & 3) * 32 + kofs + (j < 8 ? j : j + 8);       // K=128 exactly, always valid
      float v = 0.f;
      if (n < OUT_CH) v = (ks < 4) ? w2l[k * OUT_CH + n] : w2r[k * OUT_CH + n];
      __bf16 h = (__bf16)v;
      bw2hi[base + j] = h;
      bw2lo[base + j] = (__bf16)(v - (float)h);
    }
  }
}

// ------------------------------------------------------- edge scatter (mean)
// wave-per-edge: whole wave reads one coalesced source row, atomically adds
// into the target row (resolves in L2; both buffers are L2-resident).
__global__ void scatter1(const float* __restrict__ x,
                         const int* __restrict__ src, const int* __restrict__ dst,
                         float* __restrict__ agg, float* __restrict__ deg,
                         int n_edges) {
  int lane = threadIdx.x & 31;
  int w    = (blockIdx.x * blockDim.x + threadIdx.x) >> 5;
  int nw   = (gridDim.x * blockDim.x) >> 5;
  for (int e = w; e < n_edges; e += nw) {
    int s = src[e], d = dst[e];
    const float* xr = x + (size_t)s * IN_CH;
    float* ar = agg + (size_t)d * IN_CH;
    if (lane == 0) atomicAdd(&deg[d], 1.0f);
    for (int c = lane; c < IN_CH; c += 32) atomicAdd(&ar[c], xr[c]);
  }
}

__global__ void scatter2(const float* __restrict__ h1,
                         const int* __restrict__ src, const int* __restrict__ dst,
                         float* __restrict__ agg, int n_edges) {
  int lane = threadIdx.x & 31;
  int w    = (blockIdx.x * blockDim.x + threadIdx.x) >> 5;
  int nw   = (gridDim.x * blockDim.x) >> 5;
  for (int e = w; e < n_edges; e += nw) {
    int s = src[e], d = dst[e];
    const float* hr = h1 + (size_t)s * HID_CH;
    float* ar = agg + (size_t)d * HID_CH;
    for (int c = lane; c < HID_CH; c += 32) atomicAdd(&ar[c], hr[c]);
  }
}

__global__ void invdeg_k(float* deg, int n) {
  int i = blockIdx.x * blockDim.x + threadIdx.x;
  if (i < n) deg[i] = 1.0f / fmaxf(deg[i], 1.0f);
}

// ------------------------------------------------------------- layer-1 GEMM
// h1[m, 0:128] = (agg1[m]/deg) @ W1l + b1l + x[m] @ W1r
// block = 8 waves; wave w owns output cols [16w, 16w+16); block owns 16 rows.
// A fragments are built ONCE per block (wave w builds k-step w) into LDS,
// then all 8 waves read them back — 8x less global A traffic + split VALU.
__global__ void __launch_bounds__(256)
gemm1(const float* __restrict__ agg1, const float* __restrict__ x,
      const float* __restrict__ invdeg, const float* __restrict__ b1l,
      const __bf16* __restrict__ bhiW, const __bf16* __restrict__ bloW,
      float* __restrict__ h1, int n_nodes) {
  __shared__ v16bf aHi[8 * 32];
  __shared__ v16bf aLo[8 * 32];
  int tid = threadIdx.x;
  int lane = tid & 31, wave = tid >> 5;
  int half = lane >> 4, mrow = lane & 15;
  int m0 = blockIdx.x * 16;
  int m  = m0 + mrow;
  bool mv = m < n_nodes;
  int mc = mv ? m : 0;
  const float* aggrow = agg1 + (size_t)mc * IN_CH;
  const float* xrow   = x    + (size_t)mc * IN_CH;
  float inv = mv ? invdeg[m] : 0.f;
  int klo = half * 8;

  {                                   // phase 1: wave w builds k-step w
    int ks = wave;
    int kb = (ks & 3) * 32 + klo;
    const float* srow = (ks < 4) ? aggrow : xrow;
    float scl = (ks < 4) ? inv : (mv ? 1.f : 0.f);
    v16bf hi, lo;
    build_frag(srow, kb, IN_CH, scl, hi, lo);
    aHi[ks * 32 + lane] = hi;
    aLo[ks * 32 + lane] = lo;
  }
  __syncthreads();

  v8f c = {};                         // phase 2: 8 k-steps x 3 split-WMMAs
#pragma unroll
  for (int ks = 0; ks < 8; ++ks) {
    v16bf ah = aHi[ks * 32 + lane];
    v16bf al = aLo[ks * 32 + lane];
    size_t bb = ((size_t)(wave * 8 + ks) * 32 + lane) * 16;
    v16bf bh = *(const v16bf*)(bhiW + bb);
    v16bf bl = *(const v16bf*)(bloW + bb);
    c = wmma_bf16(ah, bh, c);
    c = wmma_bf16(ah, bl, c);
    c = wmma_bf16(al, bh, c);
  }

  int n = wave * 16 + mrow;           // C/D layout: all 8 VGPRs share this col
  float bias = b1l[n];
#pragma unroll
  for (int r = 0; r < 8; ++r) {
    int mo = m0 + r + half * 8;
    if (mo < n_nodes) h1[(size_t)mo * HID_CH + n] = c[r] + bias;
  }
}

// ------------------------------------------- layer-2 GEMM + log_softmax
// out[m, 0:47] = logsoftmax((agg2[m]/deg) @ W2l + b2l + h1[m] @ W2r)
// block = 4 waves: all 4 build A fragments (2 k-steps each) into LDS,
// waves 0..2 compute the 48 padded cols; LDS tile for row softmax.
__global__ void __launch_bounds__(128)
gemm2(const float* __restrict__ agg2, const float* __restrict__ h1,
      const float* __restrict__ invdeg, const float* __restrict__ b2l,
      const __bf16* __restrict__ bhiW, const __bf16* __restrict__ bloW,
      float* __restrict__ out, const int* __restrict__ orig_size, int n_nodes) {
  __shared__ v16bf aHi[8 * 32];
  __shared__ v16bf aLo[8 * 32];
  __shared__ float tile[16 * 49];
  int tid = threadIdx.x;
  int lane = tid & 31, wave = tid >> 5;
  int half = lane >> 4, mrow = lane & 15;
  int m0 = blockIdx.x * 16;
  int m  = m0 + mrow;
  bool mv = m < n_nodes;
  int mc = mv ? m : 0;
  const float* aggrow = agg2 + (size_t)mc * HID_CH;
  const float* hrow   = h1   + (size_t)mc * HID_CH;
  float inv = mv ? invdeg[m] : 0.f;
  int klo = half * 8;

#pragma unroll
  for (int i = 0; i < 2; ++i) {       // phase 1: wave w builds k-steps w, w+4
    int ks = wave + i * 4;
    int kb = (ks & 3) * 32 + klo;     // K=128 exactly, always in range
    const float* srow = (ks < 4) ? aggrow : hrow;
    float scl = (ks < 4) ? inv : (mv ? 1.f : 0.f);
    v16bf hi, lo;
    build_frag(srow, kb, HID_CH, scl, hi, lo);
    aHi[ks * 32 + lane] = hi;
    aLo[ks * 32 + lane] = lo;
  }
  __syncthreads();

  if (wave < 3) {                     // phase 2
    v8f c = {};
#pragma unroll
    for (int ks = 0; ks < 8; ++ks) {
      v16bf ah = aHi[ks * 32 + lane];
      v16bf al = aLo[ks * 32 + lane];
      size_t bb = ((size_t)(wave * 8 + ks) * 32 + lane) * 16;
      v16bf bh = *(const v16bf*)(bhiW + bb);
      v16bf bl = *(const v16bf*)(bloW + bb);
      c = wmma_bf16(ah, bh, c);
      c = wmma_bf16(ah, bl, c);
      c = wmma_bf16(al, bh, c);
    }
    int n = wave * 16 + mrow;
    float bias = (n < OUT_CH) ? b2l[n] : 0.f;
#pragma unroll
    for (int r = 0; r < 8; ++r)
      tile[(r + half * 8) * 49 + n] = c[r] + bias;
  }
  __syncthreads();

  if (tid < 16) {                     // phase 3: row-wise log_softmax
    int gm = m0 + tid;
    if (gm < orig_size[0]) {
      const float* row = tile + tid * 49;
      float mx = row[0];
      for (int j = 1; j < OUT_CH; ++j) mx = fmaxf(mx, row[j]);
      float s = 0.f;
      for (int j = 0; j < OUT_CH; ++j) s += expf(row[j] - mx);
      float ls = logf(s) + mx;
      float* orow = out + (size_t)gm * OUT_CH;
      for (int j = 0; j < OUT_CH; ++j) orow[j] = row[j] - ls;
    }
  }
}

// --------------------------------------------------------------- launcher

extern "C" void kernel_launch(void* const* d_in, const int* in_sizes, int n_in,
                              void* d_out, int out_size, void* d_ws, size_t ws_size,
                              hipStream_t stream) {
  (void)n_in; (void)out_size; (void)ws_size;
  const float* x   = (const float*)d_in[0];
  const int* eidx  = (const int*)d_in[1];
  const int* orig  = (const int*)d_in[2];
  const float* w1l = (const float*)d_in[3];
  const float* b1l = (const float*)d_in[4];
  const float* w1r = (const float*)d_in[5];
  const float* w2l = (const float*)d_in[6];
  const float* b2l = (const float*)d_in[7];
  const float* w2r = (const float*)d_in[8];

  int n_nodes = in_sizes[0] / IN_CH;
  int n_edges = in_sizes[1] / 2;
  const int* src = eidx;
  const int* dst = eidx + n_edges;

  // carve workspace (256B aligned regions)
  char* ws = (char*)d_ws;
  size_t off = 0;
  auto carve = [&](size_t bytes) -> void* {
    void* p = ws + off;
    off = (off + bytes + 255) & ~(size_t)255;
    return p;
  };
  float* deg  = (float*)carve((size_t)n_nodes * sizeof(float));
  float* agg1 = (float*)carve((size_t)n_nodes * IN_CH  * sizeof(float));
  float* agg2 = (float*)carve((size_t)n_nodes * HID_CH * sizeof(float));
  size_t zero_bytes = off;                       // deg + agg1 + agg2 need zeroing
  float*  h1    = (float*)carve((size_t)n_nodes * HID_CH * sizeof(float));
  __bf16* bw1hi = (__bf16*)carve((size_t)8 * 8 * 32 * 16 * 2);
  __bf16* bw1lo = (__bf16*)carve((size_t)8 * 8 * 32 * 16 * 2);
  __bf16* bw2hi = (__bf16*)carve((size_t)3 * 8 * 32 * 16 * 2);
  __bf16* bw2lo = (__bf16*)carve((size_t)3 * 8 * 32 * 16 * 2);

  hipMemsetAsync(d_ws, 0, zero_bytes, stream);

  prep_weights<<<(8*8*32 + 3*8*32 + 255) / 256, 256, 0, stream>>>(
      w1l, w1r, w2l, w2r, bw1hi, bw1lo, bw2hi, bw2lo);

  scatter1<<<2048, 256, 0, stream>>>(x, src, dst, agg1, deg, n_edges);
  invdeg_k<<<(n_nodes + 255) / 256, 256, 0, stream>>>(deg, n_nodes);

  int mtiles = (n_nodes + 15) / 16;
  gemm1<<<mtiles, 256, 0, stream>>>(agg1, x, deg, b1l, bw1hi, bw1lo, h1, n_nodes);

  scatter2<<<2048, 256, 0, stream>>>(h1, src, dst, agg2, n_edges);

  gemm2<<<mtiles, 128, 0, stream>>>(agg2, h1, deg, b2l, bw2hi, bw2lo,
                                    (float*)d_out, orig, n_nodes);
}
